// Attention_24781961298426
// MI455X (gfx1250) — matrix operand order repaired and checked by reading, not compiled, not run on hardware
//
#include <hip/hip_runtime.h>
#include <hip/hip_bf16.h>
#include <math.h>

// ---------------------------------------------------------------------------
// Problem constants (match reference)
// ---------------------------------------------------------------------------
constexpr int kB   = 2;
constexpr int kS   = 2048;
constexpr int kD   = 2048;
constexpr int kH   = 32;
constexpr int kKVH = 4;
constexpr int kHD  = 64;

typedef __bf16 bf16_t;
typedef __attribute__((ext_vector_type(16))) __bf16 v16bf;
typedef __attribute__((ext_vector_type(8)))  __bf16 v8bf;
typedef __attribute__((ext_vector_type(8)))  float  v8f;

// ---------------------------------------------------------------------------
// WMMA fragment helpers (wave32; C/D layout per CDNA5 ISA 7.12.2; A/B use contiguous k order)
//
// A (16x32 bf16):  lane L holds row m = L%16; lane group g = L/16.
//   fragment elems a[0..15] = K 16g..16g+15 (same contiguous order as B)
// B (32x16 bf16):  lane L holds col n = L%16; elems b[0..15] = K 16g..16g+15
// C/D (16x16 f32): lane L holds col n = L%16; VGPR v: row m = v + 8*g
//
// All loads below are two contiguous 16-byte vector loads per lane.
// ---------------------------------------------------------------------------

__device__ inline v16bf cat8(v8bf lo, v8bf hi) {
    return __builtin_shufflevector(lo, hi, 0, 1, 2, 3, 4, 5, 6, 7,
                                   8, 9, 10, 11, 12, 13, 14, 15);
}

// A fragment from row-major [rows x ld] bf16 (ld mult of 8, k0 mult of 32 -> 16B aligned)
__device__ inline v16bf load_a_frag(const bf16_t* p, int ld, int row0, int k0) {
    int lane = threadIdx.x & 31;
    int m = lane & 15, g = lane >> 4;
    const bf16_t* r = p + (size_t)(row0 + m) * ld + k0 + 16 * g;
    v8bf lo = *(const v8bf*)(r);        // K = 16g .. 16g+7
    v8bf hi = *(const v8bf*)(r + 8);    // K = 16g+8 .. 16g+15
    return cat8(lo, hi);
}

// B fragment where the logical B(k,n) = src[col0+n][k0+k] for a row-major
// [cols x ld] source (transposed operand).  Contiguous 32B per lane.
__device__ inline v16bf load_bT_frag(const bf16_t* p, int ld, int k0, int col0) {
    int lane = threadIdx.x & 31;
    int n = lane & 15, g = lane >> 4;
    const bf16_t* r = p + (size_t)(col0 + n) * ld + k0 + 16 * g;
    v8bf lo = *(const v8bf*)(r);
    v8bf hi = *(const v8bf*)(r + 8);
    return cat8(lo, hi);
}

__device__ inline void store_c_frag(float* p, int ld, int row0, int col0, v8f c) {
    int lane = threadIdx.x & 31;
    int n = lane & 15, g = lane >> 4;
#pragma unroll
    for (int v = 0; v < 8; ++v) {
        p[(size_t)(row0 + v + 8 * g) * ld + col0 + n] = c[v];
    }
}

// ---------------------------------------------------------------------------
// Kernel 1a: fp32 -> bf16 conversion (row-major preserved; for x)
// ---------------------------------------------------------------------------
__global__ void cvt_f32_bf16_kernel(const float* __restrict__ src,
                                    bf16_t* __restrict__ dst, int n) {
    int i = blockIdx.x * blockDim.x + threadIdx.x;
    if (i < n) dst[i] = (bf16_t)src[i];
}

// ---------------------------------------------------------------------------
// Kernel 1b: fp32 [R x C] -> bf16 transposed [C x R]  (for weights)
// ---------------------------------------------------------------------------
__global__ void cvt_transpose_kernel(const float* __restrict__ src,
                                     bf16_t* __restrict__ dst, int R, int C) {
    int i = blockIdx.x * blockDim.x + threadIdx.x;
    if (i >= R * C) return;
    int r = i / C, c = i - r * C;
    dst[(size_t)c * R + r] = (bf16_t)src[i];
}

// ---------------------------------------------------------------------------
// Kernel 2: bf16 GEMM  C[M,N](f32) = A[M,K](bf16) @ B[K,N](bf16)
//   B supplied TRANSPOSED: Bt [N x K] row-major.
// Block = 128 threads = 4 waves covering a 64x64 C tile (wave: 16x64).
// The 64x32 Bt tile for each k-step is staged in LDS with async copies
// (global_load_async_to_lds_b128), double-buffered, shared by all 4 waves.
// ---------------------------------------------------------------------------
__global__ void gemm_bf16_kernel(const bf16_t* __restrict__ A,
                                 const bf16_t* __restrict__ Bt,
                                 float* __restrict__ C,
                                 int M, int N, int K) {
    __shared__ alignas(16) bf16_t btile[2][64 * 32];   // [n-row][k] per buffer

    int tid  = threadIdx.x;
    int wave = tid >> 5;                               // 0..3
    int row0 = blockIdx.y * 64 + wave * 16;
    int col0 = blockIdx.x * 64;

    // Each thread asynchronously copies two 16B chunks per k-step.
    // 256 chunks of 8 bf16: chunk c -> n-row = c>>2, quarter = c&3.
    auto issue_stage = [&](int k0, int buf) {
#pragma unroll
        for (int j = 0; j < 2; ++j) {
            int c = tid * 2 + j;
            int nrow = c >> 2, quart = c & 3;
            const bf16_t* src = Bt + (size_t)(col0 + nrow) * K + k0 + quart * 8;
            unsigned dst = (unsigned)(uintptr_t)(&btile[buf][nrow * 32 + quart * 8]);
            asm volatile("global_load_async_to_lds_b128 %0, %1, off"
                         :: "v"(dst), "v"(src) : "memory");
        }
    };

    // B fragment straight out of the staged LDS tile (contiguous 32B / lane).
    auto load_b_lds = [&](const bf16_t* base, int ntile_col0) -> v16bf {
        int lane = threadIdx.x & 31;
        int n = lane & 15, g = lane >> 4;
        const bf16_t* r = base + (ntile_col0 + n) * 32 + 16 * g;
        v8bf lo = *(const v8bf*)(r);
        v8bf hi = *(const v8bf*)(r + 8);
        return cat8(lo, hi);
    };

    issue_stage(0, 0);

    v8f acc[4] = {};
    int nsteps = K >> 5;
    for (int s = 0; s < nsteps; ++s) {
        int k0 = s << 5;
        bool more = (s + 1) < nsteps;
        if (more) issue_stage(k0 + 32, (s + 1) & 1);

        // wait for this step's staged tile (leave the freshly issued 2 pending)
        if (more) asm volatile("s_wait_asynccnt 2" ::: "memory");
        else      asm volatile("s_wait_asynccnt 0" ::: "memory");
        __syncthreads();

        v16bf a = load_a_frag(A, K, row0, k0);
        if (more) {
            __builtin_prefetch(A + (size_t)(row0 + (tid & 15)) * K + k0 + 32, 0, 0);
        }
        const bf16_t* bbase = &btile[s & 1][0];
#pragma unroll
        for (int t = 0; t < 4; ++t) {
            v16bf b = load_b_lds(bbase, 16 * t);
            acc[t] = __builtin_amdgcn_wmma_f32_16x16x32_bf16(
                false, a, false, b, (short)0, acc[t], false, false);
        }
        __syncthreads();   // all waves done reading before this buffer is reused
    }

#pragma unroll
    for (int t = 0; t < 4; ++t) {
        store_c_frag(C, N, row0, col0 + 16 * t, acc[t]);
    }
}

// ---------------------------------------------------------------------------
// Kernel 3a: RoPE + transpose + bf16 convert (Q and K)
//   src:  fp32 [B, S, NH, HD]   dst: bf16 [B, NH, S, HD]
//   freqs: fp32 [S, HD/2, 2] = (cos, sin)
// ---------------------------------------------------------------------------
__global__ void rope_transpose_kernel(const float* __restrict__ src,
                                      const float* __restrict__ freqs,
                                      bf16_t* __restrict__ dst,
                                      int NH, float scale) {
    int idx = blockIdx.x * blockDim.x + threadIdx.x;
    int d2 = idx & (kHD / 2 - 1);  idx >>= 5;   // HD/2 = 32
    int h  = idx % NH;             idx /= NH;
    int s  = idx % kS;             idx /= kS;
    int b  = idx;
    if (b >= kB) return;

    const float* q = src + (((size_t)(b * kS + s) * NH + h) * kHD);
    float xr = q[2 * d2];
    float xi = q[2 * d2 + 1];
    float c  = freqs[(s * (kHD / 2) + d2) * 2 + 0];
    float sn = freqs[(s * (kHD / 2) + d2) * 2 + 1];
    float outr = xr * c - xi * sn;
    float outi = xr * sn + xi * c;

    bf16_t* o = dst + (((size_t)(b * NH + h) * kS + s) * kHD);
    o[2 * d2]     = (bf16_t)(outr * scale);
    o[2 * d2 + 1] = (bf16_t)(outi * scale);
}

// ---------------------------------------------------------------------------
// Kernel 3b: V transpose + bf16 convert
//   src: fp32 [B, S, KVH, HD]  ->  dst: bf16 [B, KVH, HD, S]
// ---------------------------------------------------------------------------
__global__ void v_transpose_kernel(const float* __restrict__ src,
                                   bf16_t* __restrict__ dst) {
    int idx = blockIdx.x * blockDim.x + threadIdx.x;
    int d = idx & (kHD - 1);  idx >>= 6;
    int h = idx % kKVH;       idx /= kKVH;
    int s = idx % kS;         idx /= kS;
    int b = idx;
    if (b >= kB) return;
    float v = src[(((size_t)(b * kS + s) * kKVH + h) * kHD) + d];
    dst[(((size_t)(b * kKVH + h) * kHD + d) * kS) + s] = (bf16_t)v;
}

// ---------------------------------------------------------------------------
// Kernel 4: flash attention, one wave per (b, h, 16-query tile).
//   Qt: bf16 [B, H,   S, HD]  (pre-scaled by 1/sqrt(HD))
//   Kt: bf16 [B, KVH, S, HD]
//   Vt: bf16 [B, KVH, HD, S]  (transposed -> contiguous B fragments)
//   Obf: bf16 [B*S, H*HD]
// ---------------------------------------------------------------------------
__global__ void attn_fwd_kernel(const bf16_t* __restrict__ Qt,
                                const bf16_t* __restrict__ Kt,
                                const bf16_t* __restrict__ Vt,
                                bf16_t* __restrict__ Obf) {
    __shared__ alignas(16) bf16_t pShared[16 * 32];   // 16x32 P tile, row-major

    int q_tile = blockIdx.x;
    int bh = blockIdx.y;                  // b*H + h
    int b = bh / kH, h = bh % kH;
    int kvh = h / (kH / kKVH);
    int q0 = q_tile * 16;

    const bf16_t* Qp = Qt + (size_t)(b * kH + h) * kS * kHD;
    const bf16_t* Kp = Kt + (size_t)(b * kKVH + kvh) * kS * kHD;
    const bf16_t* Vp = Vt + (size_t)(b * kKVH + kvh) * kHD * kS;

    int lane = threadIdx.x & 31;
    int nn = lane & 15, g = lane >> 4;

    v16bf qa0 = load_a_frag(Qp, kHD, q0, 0);
    v16bf qa1 = load_a_frag(Qp, kHD, q0, 32);

    v8f o[4] = {};
    float mrun[8], lrun[8];
#pragma unroll
    for (int i = 0; i < 8; ++i) { mrun[i] = -3.0e38f; lrun[i] = 0.0f; }

    int nchunks = (q0 + 16 + 31) >> 5;    // 32-key chunks needed for causal
    for (int c = 0; c < nchunks; ++c) {
        int key0 = c * 32;

        v8f s0 = {}, s1 = {};
        {
            v16bf kb;
            kb = load_bT_frag(Kp, kHD, 0,  key0);
            s0 = __builtin_amdgcn_wmma_f32_16x16x32_bf16(false, qa0, false, kb, (short)0, s0, false, false);
            kb = load_bT_frag(Kp, kHD, 32, key0);
            s0 = __builtin_amdgcn_wmma_f32_16x16x32_bf16(false, qa1, false, kb, (short)0, s0, false, false);
            kb = load_bT_frag(Kp, kHD, 0,  key0 + 16);
            s1 = __builtin_amdgcn_wmma_f32_16x16x32_bf16(false, qa0, false, kb, (short)0, s1, false, false);
            kb = load_bT_frag(Kp, kHD, 32, key0 + 16);
            s1 = __builtin_amdgcn_wmma_f32_16x16x32_bf16(false, qa1, false, kb, (short)0, s1, false, false);
        }

        // causal mask + online softmax update, per output row
#pragma unroll
        for (int i = 0; i < 8; ++i) {
            int qi = q0 + i + 8 * g;
            float a0 = (key0 + nn      <= qi) ? s0[i] : -1.0e9f;
            float a1 = (key0 + 16 + nn <= qi) ? s1[i] : -1.0e9f;

            float mt = fmaxf(a0, a1);
#pragma unroll
            for (int off = 1; off < 16; off <<= 1)
                mt = fmaxf(mt, __shfl_xor(mt, off, 32));

            float mnew  = fmaxf(mrun[i], mt);
            float alpha = __expf(mrun[i] - mnew);
            float p0 = __expf(a0 - mnew);
            float p1 = __expf(a1 - mnew);
            float rs = p0 + p1;
#pragma unroll
            for (int off = 1; off < 16; off <<= 1)
                rs += __shfl_xor(rs, off, 32);

            lrun[i] = lrun[i] * alpha + rs;
            mrun[i] = mnew;
#pragma unroll
            for (int t = 0; t < 4; ++t) o[t][i] *= alpha;

            int row = i + 8 * g;
            pShared[row * 32 + nn]      = (bf16_t)p0;
            pShared[row * 32 + 16 + nn] = (bf16_t)p1;
        }
        __syncthreads();

        // P(16x32) @ V(32x64): P from LDS as A fragment, V transposed source
        v16bf pa = load_a_frag(pShared, 32, 0, 0);
#pragma unroll
        for (int t = 0; t < 4; ++t) {
            v16bf vb = load_bT_frag(Vp, kS, key0, 16 * t);
            o[t] = __builtin_amdgcn_wmma_f32_16x16x32_bf16(
                false, pa, false, vb, (short)0, o[t], false, false);
        }
        __syncthreads();
    }

    // normalize + store bf16 attention output [B*S, H*HD]
#pragma unroll
    for (int i = 0; i < 8; ++i) {
        float inv = 1.0f / lrun[i];
        int row = q0 + i + 8 * g;
#pragma unroll
        for (int t = 0; t < 4; ++t) {
            Obf[(size_t)(b * kS + row) * (kH * kHD) + h * kHD + 16 * t + nn] =
                (bf16_t)(o[t][i] * inv);
        }
    }
}

// ---------------------------------------------------------------------------
// kernel_launch
// inputs (setup_inputs order): x, freqs_cis, mask, wq, wk, wv, wo
// ---------------------------------------------------------------------------
extern "C" void kernel_launch(void* const* d_in, const int* in_sizes, int n_in,
                              void* d_out, int out_size, void* d_ws, size_t ws_size,
                              hipStream_t stream) {
    const float* x     = (const float*)d_in[0];
    const float* freqs = (const float*)d_in[1];
    // d_in[2] = mask (unused; causal mask computed inline)
    const float* wq    = (const float*)d_in[3];
    const float* wk    = (const float*)d_in[4];
    const float* wv    = (const float*)d_in[5];
    const float* wo    = (const float*)d_in[6];
    float* out = (float*)d_out;

    const int M   = kB * kS;         // 4096
    const int nX  = M * kD;          // 8388608
    const int nWq = kD * kH * kHD;   // 4194304
    const int nWk = kD * kKVH * kHD; // 524288
    const int nQ  = M * kH * kHD;    // 8388608
    const int nK  = M * kKVH * kHD;  // 1048576

    // workspace carve-up (256B aligned)
    char* ws = (char*)d_ws;
    auto align_up = [](size_t v) { return (v + 255) & ~size_t(255); };
    size_t off = 0;
    auto take = [&](size_t bytes) { char* p = ws + off; off += align_up(bytes); return p; };

    bf16_t* xb   = (bf16_t*)take((size_t)nX  * 2);   // x, bf16 [4096,2048]
    bf16_t* wqt  = (bf16_t*)take((size_t)nWq * 2);   // wq^T  [2048,2048]
    bf16_t* wkt  = (bf16_t*)take((size_t)nWk * 2);   // wk^T  [256,2048]
    bf16_t* wvt  = (bf16_t*)take((size_t)nWk * 2);   // wv^T  [256,2048]
    bf16_t* wot  = (bf16_t*)take((size_t)nWq * 2);   // wo^T  [2048,2048]
    float*  qf   = (float* )take((size_t)nQ  * 4);
    float*  kf   = (float* )take((size_t)nK  * 4);
    float*  vf   = (float* )take((size_t)nK  * 4);
    bf16_t* qt   = (bf16_t*)take((size_t)nQ  * 2);   // [B,H,S,HD]
    bf16_t* kt   = (bf16_t*)take((size_t)nK  * 2);   // [B,KVH,S,HD]
    bf16_t* vt   = (bf16_t*)take((size_t)nK  * 2);   // [B,KVH,HD,S]
    bf16_t* attb = (bf16_t*)take((size_t)nQ  * 2);   // [B*S, H*HD]
    (void)ws_size;

    // 1) conversions (weights transposed for contiguous B fragments)
    cvt_f32_bf16_kernel<<<(nX + 255) / 256, 256, 0, stream>>>(x, xb, nX);
    cvt_transpose_kernel<<<(nWq + 255) / 256, 256, 0, stream>>>(wq, wqt, kD, kH * kHD);
    cvt_transpose_kernel<<<(nWk + 255) / 256, 256, 0, stream>>>(wk, wkt, kD, kKVH * kHD);
    cvt_transpose_kernel<<<(nWk + 255) / 256, 256, 0, stream>>>(wv, wvt, kD, kKVH * kHD);
    cvt_transpose_kernel<<<(nWq + 255) / 256, 256, 0, stream>>>(wo, wot, kH * kHD, kD);

    // 2) projections:  Q = x@wq  (N=2048),  K = x@wk, V = x@wv  (N=256)
    {
        dim3 blk(128);
        dim3 grdQ(kH * kHD / 64, M / 64);      // (32, 64)
        dim3 grdK(kKVH * kHD / 64, M / 64);    // (4, 64)
        gemm_bf16_kernel<<<grdQ, blk, 0, stream>>>(xb, wqt, qf, M, kH * kHD,  kD);
        gemm_bf16_kernel<<<grdK, blk, 0, stream>>>(xb, wkt, kf, M, kKVH * kHD, kD);
        gemm_bf16_kernel<<<grdK, blk, 0, stream>>>(xb, wvt, vf, M, kKVH * kHD, kD);
    }

    // 3) RoPE + transposes (Q pre-scaled by 1/sqrt(HD) = 0.125)
    {
        int pq = kB * kS * kH   * (kHD / 2);
        int pk = kB * kS * kKVH * (kHD / 2);
        int pv = kB * kS * kKVH * kHD;
        rope_transpose_kernel<<<(pq + 255) / 256, 256, 0, stream>>>(qf, freqs, qt, kH,   0.125f);
        rope_transpose_kernel<<<(pk + 255) / 256, 256, 0, stream>>>(kf, freqs, kt, kKVH, 1.0f);
        v_transpose_kernel<<<(pv + 255) / 256, 256, 0, stream>>>(vf, vt);
    }

    // 4) flash attention: one wave per (b, h, 16-query tile)
    {
        dim3 blk(32);
        dim3 grd(kS / 16, kB * kH);            // (128, 64)
        attn_fwd_kernel<<<grd, blk, 0, stream>>>(qt, kt, vt, attb);
    }

    // 5) output projection: out = attn @ wo  (fp32 result)
    {
        dim3 blk(128);
        dim3 grd(kD / 64, M / 64);             // (32, 64)
        gemm_bf16_kernel<<<grd, blk, 0, stream>>>(attb, wot, out, M, kD, kH * kHD);
    }
}
